// GNN_20237885899323
// MI455X (gfx1250) — matrix operand order repaired and checked by reading, not compiled, not run on hardware
//
#include <hip/hip_runtime.h>
#include <hip/hip_bf16.h>

// ---------------------------------------------------------------------------
// GCN pipeline for MI455X (gfx1250, wave32, WMMA).
//   h0  = relu(x @ w_node + b_node)                      [N,32]  (f16 for WMMA)
//   xw  = h0 @ w1  (WMMA f32_16x16x32_f16)               [N,64]
//   acc = self-loop init + edge scatter (L2 atomics)     [N,64]
//   h1  = relu(acc + b1)  (f16)                          [N,64]
//   xw  = h1 @ w2  (WMMA, K=64)                          [N,64]
//   ... scatter ... h2 = relu(acc + b2)                  [N,64]  (f32)
//   mean-pool per graph, tiny MLP head -> out            [G,1]
// The edge_attr/w_edge branch is dead w.r.t. the output and is skipped.
// Roofline: dense GEMMs ~1.2 GFLOP (negligible); cost is the edge scatter,
// whose working set (xw 25.6MB + acc 25.6MB + dis 0.4MB) is L2-resident
// (192MB L2), so it is L2-atomic bound, not HBM bound.
// ---------------------------------------------------------------------------

typedef __attribute__((ext_vector_type(16))) _Float16 v16h;
typedef __attribute__((ext_vector_type(8)))  _Float16 v8h;
typedef __attribute__((ext_vector_type(8)))  float    v8f;

// ---- weight conversion (f32 -> f16) ---------------------------------------
__global__ void gnn_cvt_weights(const float* __restrict__ w1,
                                const float* __restrict__ w2,
                                _Float16* __restrict__ w1h,
                                _Float16* __restrict__ w2h) {
    int i = blockIdx.x * blockDim.x + threadIdx.x;
    if (i < 32 * 64) w1h[i] = (_Float16)w1[i];
    if (i < 64 * 64) w2h[i] = (_Float16)w2[i];
}

// ---- init: deg=1 (self loop), sums=0, cnts=0 ------------------------------
__global__ void gnn_init(float* __restrict__ deg, float* __restrict__ sums,
                         float* __restrict__ cnts, int N, int G) {
    int i = blockIdx.x * blockDim.x + threadIdx.x;
    if (i < N) deg[i] = 1.0f;
    if (i < G * 64) sums[i] = 0.0f;
    if (i < G) cnts[i] = 0.0f;
}

// ---- node embedding: h0 = relu(x @ w_node + b_node), stored f16 -----------
__global__ void gnn_node_embed(const float* __restrict__ x,
                               const float* __restrict__ wn,
                               const float* __restrict__ bn,
                               _Float16* __restrict__ h0, int N) {
    long long i = (long long)blockIdx.x * blockDim.x + threadIdx.x;
    if (i >= (long long)N * 32) return;
    int n = (int)(i >> 5), j = (int)(i & 31);
    float v = x[n * 2 + 0] * wn[j] + x[n * 2 + 1] * wn[32 + j] + bn[j];
    h0[(size_t)n * 32 + j] = (_Float16)fmaxf(v, 0.0f);
}

// ---- degree accumulation over in-edges ------------------------------------
__global__ void gnn_deg(const int* __restrict__ ei, float* __restrict__ deg, int E) {
    int e = blockIdx.x * blockDim.x + threadIdx.x;
    if (e >= E) return;
    atomicAdd(&deg[ei[(long long)E + e]], 1.0f);   // dst = ei[1][e]
}

__global__ void gnn_rsqrt(float* __restrict__ d, int N) {
    int i = blockIdx.x * blockDim.x + threadIdx.x;
    if (i < N) d[i] = rsqrtf(d[i]);                // deg >= 1 always
}

// ---- WMMA GEMM: C[M,64] = A[M,K](f16) @ B[K,64](f16), K in {32,64} --------
template <int K>
__global__ void gnn_gemm_wmma(const _Float16* __restrict__ A,
                              const _Float16* __restrict__ B,
                              float* __restrict__ C, int M) {
    const int lane = threadIdx.x & 31;
    const long long wave =
        (long long)blockIdx.x * (blockDim.x >> 5) + (threadIdx.x >> 5);
    const long long tiles = (M + 15) >> 4;
    if (wave >= tiles) return;                 // wave-uniform: EXEC stays full
    const int row0 = (int)(wave << 4);
    const int g = lane >> 4;                   // K-half selector for A frag
    const int m = lane & 15;
    int arow = row0 + m;
    if (arow >= M) arow = M - 1;               // clamp loads; stores masked below
    const _Float16* aptr = A + (size_t)arow * K;

    v8f acc[4] = {};
#pragma unroll
    for (int ks = 0; ks < K; ks += 32) {
        // A 16x32 fragment: elems 0..7 <- K=ks+g*8.., elems 8..15 <- K=ks+16+g*8..
        v8h lo = *(const v8h*)(aptr + ks + g * 8);
        v8h hi = *(const v8h*)(aptr + ks + 16 + g * 8);
        v16h a;
#pragma unroll
        for (int e = 0; e < 8; ++e) { a[e] = lo[e]; a[e + 8] = hi[e]; }
        // B 32x16 fragment per N-tile: lane holds column n=m, K same order as A
        const _Float16* bptr = B + (size_t)ks * 64 + m;
#pragma unroll
        for (int t = 0; t < 4; ++t) {
            v16h b;
#pragma unroll
            for (int e = 0; e < 8; ++e) {
                b[e] = bptr[(size_t)(g * 8 + e) * 64 + t * 16];
                b[e + 8] = bptr[(size_t)(16 + g * 8 + e) * 64 + t * 16];
            }
            acc[t] = __builtin_amdgcn_wmma_f32_16x16x32_f16(
                false, a, false, b, (short)0, acc[t], false, false);
        }
    }
    // D layout: VGPR r, lane l -> (m = r + 8*(l>=16), n = l&15)
    const int ncol = lane & 15;
    const int mhalf = lane >> 4;
    float* crow = C + (size_t)(row0 + 8 * mhalf) * 64 + ncol;
    if (row0 + 16 <= M) {
        // Fast path (all tiles except at most one ragged tail): straight-line
        // unguarded stores -> no per-element exec-mask manipulation.
#pragma unroll
        for (int t = 0; t < 4; ++t)
#pragma unroll
            for (int r = 0; r < 8; ++r)
                crow[(size_t)r * 64 + t * 16] = acc[t][r];
    } else {
#pragma unroll
        for (int t = 0; t < 4; ++t)
#pragma unroll
            for (int r = 0; r < 8; ++r) {
                int rr = row0 + r + 8 * mhalf;
                if (rr < M) C[(size_t)rr * 64 + t * 16 + ncol] = acc[t][r];
            }
    }
}

// ---- self-loop init: acc = xw * dis^2 -------------------------------------
__global__ void gnn_selfloop(const float* __restrict__ xw,
                             const float* __restrict__ dis,
                             float* __restrict__ acc, int N) {
    long long i = (long long)blockIdx.x * blockDim.x + threadIdx.x;
    if (i >= (long long)N * 64) return;
    int n = (int)(i >> 6);
    float d = dis[n];
    acc[i] = xw[i] * d * d;
}

// ---- edge scatter: acc[dst] += xw[src] * dis[src]*dis[dst] ----------------
// 64 threads per edge; xw/acc are L2-resident (25.6 MB each << 192 MB L2).
__global__ void gnn_scatter(const int* __restrict__ ei,
                            const float* __restrict__ xw,
                            const float* __restrict__ dis,
                            float* __restrict__ acc, int E) {
    long long tid = (long long)blockIdx.x * blockDim.x + threadIdx.x;
    long long e = tid >> 6;
    if (e >= E) return;
    int j = (int)(tid & 63);
    int src = ei[e];
    int dst = ei[(long long)E + e];
    float norm = dis[src] * dis[dst];
    atomicAdd(&acc[(size_t)dst * 64 + j], xw[(size_t)src * 64 + j] * norm);
}

// ---- finalize: relu(acc + b) -> f16 (feeds next WMMA) or f32 --------------
__global__ void gnn_finalize_f16(const float* __restrict__ acc,
                                 const float* __restrict__ b,
                                 _Float16* __restrict__ h, int N) {
    long long i = (long long)blockIdx.x * blockDim.x + threadIdx.x;
    if (i >= (long long)N * 64) return;
    h[i] = (_Float16)fmaxf(acc[i] + b[i & 63], 0.0f);
}

__global__ void gnn_finalize_f32(const float* __restrict__ acc,
                                 const float* __restrict__ b,
                                 float* __restrict__ h, int N) {
    long long i = (long long)blockIdx.x * blockDim.x + threadIdx.x;
    if (i >= (long long)N * 64) return;
    h[i] = fmaxf(acc[i] + b[i & 63], 0.0f);
}

// ---- mean pool accumulation -----------------------------------------------
__global__ void gnn_pool(const float* __restrict__ h,
                         const int* __restrict__ batch,
                         float* __restrict__ sums, float* __restrict__ cnts,
                         int N) {
    long long i = (long long)blockIdx.x * blockDim.x + threadIdx.x;
    if (i >= (long long)N * 64) return;
    int n = (int)(i >> 6), j = (int)(i & 63);
    int g = batch[n];
    atomicAdd(&sums[(size_t)g * 64 + j], h[i]);
    if (j == 0) atomicAdd(&cnts[g], 1.0f);
}

// ---- MLP head: out = (relu(pooled@wf1+bf1)) @ wf2 + bf2 -------------------
__global__ void gnn_head(const float* __restrict__ sums,
                         const float* __restrict__ cnts,
                         const float* __restrict__ wf1,
                         const float* __restrict__ bf1,
                         const float* __restrict__ wf2,
                         const float* __restrict__ bf2,
                         float* __restrict__ out, int G) {
    int g = blockIdx.x * blockDim.x + threadIdx.x;
    if (g >= G) return;
    float inv = 1.0f / fmaxf(cnts[g], 1.0f);
    float pooled[64];
#pragma unroll
    for (int k = 0; k < 64; ++k) pooled[k] = sums[(size_t)g * 64 + k] * inv;
    float o = bf2[0];
    for (int j = 0; j < 32; ++j) {
        float a = bf1[j];
#pragma unroll
        for (int k = 0; k < 64; ++k) a += pooled[k] * wf1[k * 32 + j];
        o += fmaxf(a, 0.0f) * wf2[j];
    }
    out[g] = o;
}

// ---------------------------------------------------------------------------
extern "C" void kernel_launch(void* const* d_in, const int* in_sizes, int n_in,
                              void* d_out, int out_size, void* d_ws, size_t ws_size,
                              hipStream_t stream) {
    (void)n_in; (void)ws_size;
    const float* x      = (const float*)d_in[0];
    // d_in[1] edge_attr: dead w.r.t. output
    const int*   ei     = (const int*)d_in[2];
    const int*   batch  = (const int*)d_in[3];
    const float* w_node = (const float*)d_in[4];
    const float* b_node = (const float*)d_in[5];
    // d_in[6..7] w_edge/b_edge: dead
    const float* w1  = (const float*)d_in[8];
    const float* b1  = (const float*)d_in[9];
    const float* w2  = (const float*)d_in[10];
    const float* b2  = (const float*)d_in[11];
    const float* wf1 = (const float*)d_in[12];
    const float* bf1 = (const float*)d_in[13];
    const float* wf2 = (const float*)d_in[14];
    const float* bf2 = (const float*)d_in[15];

    const int N = in_sizes[0] / 2;   // 100000
    const int E = in_sizes[2] / 2;   // 3200000
    const int G = out_size;          // 1024

    // workspace carve (256B-aligned chunks; ~71 MB total)
    char* ws = (char*)d_ws;
    size_t off = 0;
    auto carve = [&](size_t bytes) -> char* {
        char* p = ws + off;
        off = (off + bytes + 255) & ~(size_t)255;
        return p;
    };
    float*    dis  = (float*)carve((size_t)N * 4);            // deg -> rsqrt(deg)
    _Float16* h0h  = (_Float16*)carve((size_t)N * 32 * 2);
    _Float16* h1h  = (_Float16*)carve((size_t)N * 64 * 2);
    float*    xw   = (float*)carve((size_t)N * 64 * 4);       // also reused for h2
    float*    acc  = (float*)carve((size_t)N * 64 * 4);
    float*    sums = (float*)carve((size_t)G * 64 * 4);
    float*    cnts = (float*)carve((size_t)G * 4);
    _Float16* w1h  = (_Float16*)carve(32 * 64 * 2);
    _Float16* w2h  = (_Float16*)carve(64 * 64 * 2);

    const int TB = 256;  // 8 wave32 waves per block
    auto blocks = [](long long work, int tb) -> unsigned {
        return (unsigned)((work + tb - 1) / tb);
    };

    gnn_cvt_weights<<<blocks(64 * 64, TB), TB, 0, stream>>>(w1, w2, w1h, w2h);
    gnn_init<<<blocks(N, TB), TB, 0, stream>>>(dis, sums, cnts, N, G);
    gnn_node_embed<<<blocks((long long)N * 32, TB), TB, 0, stream>>>(x, w_node, b_node, h0h, N);
    gnn_deg<<<blocks(E, TB), TB, 0, stream>>>(ei, dis, E);
    gnn_rsqrt<<<blocks(N, TB), TB, 0, stream>>>(dis, N);

    const long long tiles = (N + 15) / 16;            // one wave per 16-node tile
    const unsigned gemmBlocks = blocks(tiles * 32, TB);

    // ---- GCN layer 1 ----
    gnn_gemm_wmma<32><<<gemmBlocks, TB, 0, stream>>>(h0h, w1h, xw, N);
    gnn_selfloop<<<blocks((long long)N * 64, TB), TB, 0, stream>>>(xw, dis, acc, N);
    gnn_scatter<<<blocks((long long)E * 64, TB), TB, 0, stream>>>(ei, xw, dis, acc, E);
    gnn_finalize_f16<<<blocks((long long)N * 64, TB), TB, 0, stream>>>(acc, b1, h1h, N);

    // ---- GCN layer 2 ----
    gnn_gemm_wmma<64><<<gemmBlocks, TB, 0, stream>>>(h1h, w2h, xw, N);
    gnn_selfloop<<<blocks((long long)N * 64, TB), TB, 0, stream>>>(xw, dis, acc, N);
    gnn_scatter<<<blocks((long long)E * 64, TB), TB, 0, stream>>>(ei, xw, dis, acc, E);
    gnn_finalize_f32<<<blocks((long long)N * 64, TB), TB, 0, stream>>>(acc, b2, xw, N);

    // ---- pool + head ----
    gnn_pool<<<blocks((long long)N * 64, TB), TB, 0, stream>>>(xw, batch, sums, cnts, N);
    gnn_head<<<blocks(G, TB), TB, 0, stream>>>(sums, cnts, wf1, bf1, wf2, bf2,
                                               (float*)d_out, G);
}
